// FFB_encoder_51737176047691
// MI455X (gfx1250) — matrix operand order repaired and checked by reading, not compiled, not run on hardware
//
#include <hip/hip_runtime.h>
#include <hip/hip_bf16.h>

typedef _Float16 half16 __attribute__((ext_vector_type(16)));
typedef float    float8 __attribute__((ext_vector_type(8)));

typedef unsigned int v4u_t __attribute__((ext_vector_type(4)));
typedef int          v8i_t __attribute__((ext_vector_type(8)));
typedef int          v4i_t __attribute__((ext_vector_type(4)));

#define NPTS      500000
#define TBITS     19
#define TSIZE     (1u << TBITS)
#define LPB       128                       // points per block (8 waves x 16)
#define NBLOCKS   ((NPTS + LPB - 1) / LPB)  // 3907
#define WPL       12288                     // weight halves per layer (8+16 frags * 512)
#define WSW_HALVES (4 * WPL)                // all-layer pre-swizzled weights

// TDM availability (evaluated per compilation pass; host pass sees 0 and
// parses the plain-copy fallback instead, which is all it needs).
#if defined(__has_builtin)
#  if __has_builtin(__builtin_amdgcn_tensor_load_to_lds) && \
      __has_builtin(__builtin_amdgcn_s_wait_tensorcnt)
#    define FFB_HAS_TDM 1
#  endif
#endif
#ifndef FFB_HAS_TDM
#  define FFB_HAS_TDM 0
#endif

// ---------------------------------------------------------------------------
// Prologue: convert W_hidden/W_high (f32) into f16 pre-swizzled WMMA
// B-fragment layout in workspace. Layout per layer l (WPL halves):
//   hidden frags [0,4096): frag = (n>>4)*2 + (k>>5); lane*16 + (k&15)
//   high   frags [4096,12288): same formula, Nfull=128
// ---------------------------------------------------------------------------
__global__ __launch_bounds__(256) void ffb_prep_weights(
    const float* __restrict__ W_hidden,   // [4,64,64]
    const float* __restrict__ W_high,     // [4,64,128]
    _Float16* __restrict__ wsw)           // [4*WPL]
{
    int stride = gridDim.x * 256;
    for (int e = blockIdx.x * 256 + threadIdx.x; e < 4 * WPL; e += stride) {
        int l  = e / WPL;
        int e1 = e - l * WPL;
        if (e1 < 4096) {                        // W_hidden[l]: 64x64
            int k = e1 >> 6, n = e1 & 63;
            float v = W_hidden[((size_t)l * 64 + k) * 64 + n];
            int frag = (n >> 4) * 2 + (k >> 5);
            int kk = k & 31;
            int ln = (n & 15) + 16 * (kk >> 4);
            wsw[(size_t)l * WPL + frag * 512 + ln * 16 + (kk & 15)] = (_Float16)v;
        } else {                                // W_high[l]: 64x128
            int e2 = e1 - 4096;
            int k = e2 >> 7, n = e2 & 127;
            float v = W_high[((size_t)l * 64 + k) * 128 + n];
            int frag = (n >> 4) * 2 + (k >> 5);
            int kk = k & 31;
            int ln = (n & 15) + 16 * (kk >> 4);
            wsw[(size_t)l * WPL + 4096 + frag * 512 + ln * 16 + (kk & 15)] = (_Float16)v;
        }
    }
}

// ---------------------------------------------------------------------------
// FFB encoder: hash-grid + Fourier features + SIREN MLP, WMMA f16->f32 matmuls
// ---------------------------------------------------------------------------
__global__ __launch_bounds__(256) void ffb_encoder_kernel(
    const float* __restrict__ in_pos,     // [N,3]
    const float* __restrict__ W_first,    // [3,64]
    const float* __restrict__ b_first,    // [64]
    const float* __restrict__ W_hidden,   // [4,64,64]
    const float* __restrict__ b_hidden,   // [4,64]
    const float* __restrict__ W_high,     // [4,64,128]
    const float* __restrict__ b_high,     // [4,128]
    const float* __restrict__ ffn_A,      // [4,2,64]
    const float* __restrict__ hash_table, // [4,T,2]
    const _Float16* __restrict__ wsw,     // pre-swizzled weights or nullptr
    float* __restrict__ out)              // [N,128]
{
    // ---- LDS ----
    __shared__ __align__(32) _Float16 sW[WPL];          // current layer, 24 KB
    __shared__ __align__(32) _Float16 sX[8192];         // A-frag staging, 16 KB
    __shared__ float sG[4 * LPB * 2];                   // hash-grid feats, 4 KB
    __shared__ float sPos[LPB * 3];
    __shared__ float sA[4 * 2 * 64];                    // ffn_A * sigma*2pi
    __shared__ float sBh[4 * 64];
    __shared__ float sBo[4 * 128];
    __shared__ float sWf[3 * 64];
    __shared__ float sBf[64];

    const int tid       = threadIdx.x;
    const int lane      = tid & 31;
    const int wv        = tid >> 5;
    const int blockBase = blockIdx.x * LPB;
    const int r         = lane & 15;
    const int hi        = lane >> 4;
    const int mBase     = 8 * hi;
    const int pW        = wv * 16;

    // ================= cooperative phase =================
    for (int p = tid; p < LPB; p += 256) {
        int gp = blockBase + p;
        float x = 0.f, y = 0.f, z = 0.f;
        if (gp < NPTS) {
            x = in_pos[(size_t)gp * 3 + 0];
            y = in_pos[(size_t)gp * 3 + 1];
            z = in_pos[(size_t)gp * 3 + 2];
        }
        sPos[p * 3 + 0] = x; sPos[p * 3 + 1] = y; sPos[p * 3 + 2] = z;
    }
    for (int i = tid; i < 4 * 2 * 64; i += 256) {
        int l = i >> 7;
        float sig = (float)(1 << l) * 6.2831853071795864f;
        sA[i] = ffn_A[i] * sig;
    }
    for (int i = tid; i < 4 * 64;  i += 256) sBh[i] = b_hidden[i];
    for (int i = tid; i < 4 * 128; i += 256) sBo[i] = b_high[i];
    for (int i = tid; i < 3 * 64;  i += 256) sWf[i] = W_first[i];
    for (int i = tid; i < 64;      i += 256) sBf[i] = b_first[i];
    __syncthreads();

    // ---- hash-grid trilinear interp: one (point, level) task per thread ----
    for (int task = tid; task < 4 * LPB; task += 256) {
        int l  = task >> 7;
        int p  = task & (LPB - 1);
        int gp = blockBase + p;
        float g0 = 0.f, g1 = 0.f;
        if (gp < NPTS) {
            float scale = (float)(16 << l) - 1.0f;
            float px = (sPos[p * 3 + 0] + 1.f) * 0.5f;
            float py = (sPos[p * 3 + 1] + 1.f) * 0.5f;
            float pz = (sPos[p * 3 + 2] + 1.f) * 0.5f;
            float sx = px * scale + 0.5f;
            float sy = py * scale + 0.5f;
            float sz = pz * scale + 0.5f;
            float fx = floorf(sx), fy = floorf(sy), fz = floorf(sz);
            float wx = sx - fx, wy = sy - fy, wz = sz - fz;
            unsigned c0 = (unsigned)fx, c1 = (unsigned)fy, c2 = (unsigned)fz;
            const float* tab = hash_table + (size_t)l * TSIZE * 2;
            #pragma unroll
            for (int corner = 0; corner < 8; ++corner) {
                unsigned b0 =  corner       & 1;
                unsigned b1 = (corner >> 1) & 1;
                unsigned b2 = (corner >> 2) & 1;
                unsigned h = (c0 + b0) * 1u
                           ^ (c1 + b1) * 2654435761u
                           ^ (c2 + b2) * 805459861u;
                unsigned idx = h & (TSIZE - 1u);
                float wt = (b0 ? wx : 1.f - wx) * (b1 ? wy : 1.f - wy)
                         * (b2 ? wz : 1.f - wz);
                const float* f = tab + (size_t)idx * 2;
                g0 += wt * f[0];
                g1 += wt * f[1];
            }
        }
        sG[(l * LPB + p) * 2 + 0] = g0;
        sG[(l * LPB + p) * 2 + 1] = g1;
    }
    __syncthreads();

    // ================= first SIREN layer (3 -> 64, elementwise) =============
    _Float16* xs = sX + wv * 1024;
    float xacc[4][8];
    #pragma unroll
    for (int nt = 0; nt < 4; ++nt) {
        int f = nt * 16 + r;
        float w0 = sWf[0 * 64 + f], w1 = sWf[1 * 64 + f], w2 = sWf[2 * 64 + f];
        float bb = sBf[f];
        #pragma unroll
        for (int j = 0; j < 8; ++j) {
            int p = pW + mBase + j;
            float v = sPos[p * 3 + 0] * w0 + sPos[p * 3 + 1] * w1
                    + sPos[p * 3 + 2] * w2 + bb;
            xacc[nt][j] = __sinf(30.f * v);
        }
    }
    #pragma unroll
    for (int nt = 0; nt < 4; ++nt) {
        int slot = r;
        #pragma unroll
        for (int j = 0; j < 8; ++j) {
            int m = mBase + j;
            int laneA = m + 16 * (nt & 1);
            xs[(nt >> 1) * 512 + laneA * 16 + slot] = (_Float16)xacc[nt][j];
        }
    }

    // ================= main loop over 4 grid levels =========================
    float oacc[8][8];
    #pragma unroll
    for (int nt = 0; nt < 8; ++nt)
        #pragma unroll
        for (int j = 0; j < 8; ++j) oacc[nt][j] = 0.f;

    for (int l = 0; l < 4; ++l) {
        __syncthreads();   // all waves done reading previous layer's sW

        if (wsw != nullptr) {
#if FFB_HAS_TDM
            // ---- Tensor Data Mover: DMA 24KB pre-swizzled f16 layer image
            //      from L2-resident workspace straight into LDS ----
            if (wv == 0) {
                unsigned long long ga =
                    (unsigned long long)(const void*)(wsw + (size_t)l * WPL);
                unsigned lds_addr = (unsigned)(size_t)(void*)sW;

                v4u_t g0;
                g0.x = 1u;                                   // count=1 (valid D#)
                g0.y = lds_addr;                             // lds_addr[31:0]
                g0.z = (unsigned)(ga & 0xFFFFFFFFu);         // global_addr[31:0]
                g0.w = (unsigned)((ga >> 32) & 0x01FFFFFFu)  // global_addr[56:32]
                     | (2u << 30);                           // type=2 ("image")

                v8i_t g1;
                g1[0] = (int)(1u << 16);                     // data_size=1 (2B)
                g1[1] = (int)((unsigned)WPL << 16);          // tensor_dim0[15:0]
                g1[2] = (int)(1u << 16);                     // tdim0[31:16]=0, tensor_dim1=1
                g1[3] = (int)((unsigned)WPL << 16);          // tile_dim0=WPL (1-D tile)
                g1[4] = 0;                                   // tile_dim1/2 unused
                g1[5] = (int)WPL;                            // tensor_dim0_stride
                g1[6] = 0;
                g1[7] = 0;

                v4i_t gz = {0, 0, 0, 0};
#if __clang_major__ >= 23
                v8i_t gz8 = {0, 0, 0, 0, 0, 0, 0, 0};
                __builtin_amdgcn_tensor_load_to_lds(g0, g1, gz, gz, gz8, 0);
#else
                __builtin_amdgcn_tensor_load_to_lds(g0, g1, gz, gz, 0);
#endif
                __builtin_amdgcn_s_wait_tensorcnt(0);
            }
#else
            // fallback: plain vectorized LDS copy of pre-swizzled f16 weights
            {
                const uint4* src = (const uint4*)(wsw + (size_t)l * WPL);
                uint4* dst = (uint4*)sW;
                for (int e = tid; e < WPL / 8; e += 256) dst[e] = src[e];
            }
#endif
        } else {
            // ---- fallback: cooperative f32 load + cvt + swizzle ----
            for (int e = tid; e < WPL; e += 256) {
                if (e < 4096) {
                    int k = e >> 6, n = e & 63;
                    float v = W_hidden[((size_t)l * 64 + k) * 64 + n];
                    int frag = (n >> 4) * 2 + (k >> 5);
                    int kk = k & 31;
                    int ln = (n & 15) + 16 * (kk >> 4);
                    sW[frag * 512 + ln * 16 + (kk & 15)] = (_Float16)v;
                } else {
                    int e2 = e - 4096;
                    int k = e2 >> 7, n = e2 & 127;
                    float v = W_high[((size_t)l * 64 + k) * 128 + n];
                    int frag = (n >> 4) * 2 + (k >> 5);
                    int kk = k & 31;
                    int ln = (n & 15) + 16 * (kk >> 4);
                    sW[4096 + frag * 512 + ln * 16 + (kk & 15)] = (_Float16)v;
                }
            }
        }
        __syncthreads();

        // ---- A fragments of current x (K = 0..31 and 32..63) ----
        half16 a0 = *reinterpret_cast<const half16*>(xs + lane * 16);
        half16 a1 = *reinterpret_cast<const half16*>(xs + 512 + lane * 16);

        // ---- hidden matmul (64->64) + sin + Fourier add ----
        #pragma unroll
        for (int nt = 0; nt < 4; ++nt) {
            half16 b0 = *reinterpret_cast<const half16*>(sW + (nt * 2 + 0) * 512 + lane * 16);
            half16 b1 = *reinterpret_cast<const half16*>(sW + (nt * 2 + 1) * 512 + lane * 16);
            float8 c = {0.f, 0.f, 0.f, 0.f, 0.f, 0.f, 0.f, 0.f};
            c = __builtin_amdgcn_wmma_f32_16x16x32_f16(false, a0, false, b0,
                                                       (short)0, c, false, false);
            c = __builtin_amdgcn_wmma_f32_16x16x32_f16(false, a1, false, b1,
                                                       (short)0, c, false, false);
            int f = nt * 16 + r;
            float bb  = sBh[l * 64 + f];
            float a0f = sA[(l * 2 + 0) * 64 + f];
            float a1f = sA[(l * 2 + 1) * 64 + f];
            #pragma unroll
            for (int j = 0; j < 8; ++j) {
                int p = pW + mBase + j;
                float g0 = sG[(l * LPB + p) * 2 + 0];
                float g1 = sG[(l * LPB + p) * 2 + 1];
                xacc[nt][j] = __sinf(30.f * (c[j] + bb))
                            + __sinf(g0 * a0f + g1 * a1f);
            }
        }
        // ---- restage x (wave-private; serves head now + next hidden) ----
        #pragma unroll
        for (int nt = 0; nt < 4; ++nt) {
            int slot = r;
            #pragma unroll
            for (int j = 0; j < 8; ++j) {
                int m = mBase + j;
                int laneA = m + 16 * (nt & 1);
                xs[(nt >> 1) * 512 + laneA * 16 + slot] = (_Float16)xacc[nt][j];
            }
        }
        a0 = *reinterpret_cast<const half16*>(xs + lane * 16);
        a1 = *reinterpret_cast<const half16*>(xs + 512 + lane * 16);

        // ---- head matmul (64->128) + sin, accumulate into output ----
        #pragma unroll
        for (int nt = 0; nt < 8; ++nt) {
            half16 b0 = *reinterpret_cast<const half16*>(sW + 4096 + (nt * 2 + 0) * 512 + lane * 16);
            half16 b1 = *reinterpret_cast<const half16*>(sW + 4096 + (nt * 2 + 1) * 512 + lane * 16);
            float8 c = {0.f, 0.f, 0.f, 0.f, 0.f, 0.f, 0.f, 0.f};
            c = __builtin_amdgcn_wmma_f32_16x16x32_f16(false, a0, false, b0,
                                                       (short)0, c, false, false);
            c = __builtin_amdgcn_wmma_f32_16x16x32_f16(false, a1, false, b1,
                                                       (short)0, c, false, false);
            float bb = sBo[l * 128 + nt * 16 + r];
            #pragma unroll
            for (int j = 0; j < 8; ++j)
                oacc[nt][j] += __sinf(30.f * (c[j] + bb));
        }
    }

    // ================= write output [N,128] =================
    #pragma unroll
    for (int nt = 0; nt < 8; ++nt) {
        #pragma unroll
        for (int j = 0; j < 8; ++j) {
            int gp = blockBase + pW + mBase + j;
            if (gp < NPTS)
                out[(size_t)gp * 128 + nt * 16 + r] = oacc[nt][j];
        }
    }
}

extern "C" void kernel_launch(void* const* d_in, const int* in_sizes, int n_in,
                              void* d_out, int out_size, void* d_ws, size_t ws_size,
                              hipStream_t stream) {
    (void)in_sizes; (void)n_in; (void)out_size;
    const float* in_pos     = (const float*)d_in[0];
    const float* W_first    = (const float*)d_in[1];
    const float* b_first    = (const float*)d_in[2];
    const float* W_hidden   = (const float*)d_in[3];
    const float* b_hidden   = (const float*)d_in[4];
    const float* W_high     = (const float*)d_in[5];
    const float* b_high     = (const float*)d_in[6];
    const float* ffn_A      = (const float*)d_in[7];
    const float* hash_table = (const float*)d_in[8];
    float* out = (float*)d_out;

    const size_t need = (size_t)WSW_HALVES * sizeof(_Float16);   // 96 KB
    _Float16* wsw = (ws_size >= need) ? (_Float16*)d_ws : nullptr;

    if (wsw)
        ffb_prep_weights<<<64, 256, 0, stream>>>(W_hidden, W_high, wsw);

    ffb_encoder_kernel<<<NBLOCKS, 256, 0, stream>>>(
        in_pos, W_first, b_first, W_hidden, b_hidden,
        W_high, b_high, ffn_A, hash_table, wsw, out);
}
